// AxialCrossAttention_61692910239786
// MI455X (gfx1250) — matrix operand order repaired and checked by reading, not compiled, not run on hardware
//
#include <hip/hip_runtime.h>
#include <hip/hip_bf16.h>

typedef __attribute__((ext_vector_type(16))) _Float16 v16h;
typedef __attribute__((ext_vector_type(8)))  float    v8f;

// ---------------------------------------------------------------------------
// WMMA helpers (CDNA5 / gfx1250, wave32)
// ---------------------------------------------------------------------------
__device__ __forceinline__ v8f wmma_f32(v16h a, v16h b, v8f c) {
  // D = A(16x32 f16) x B(32x16 f16) + C(16x16 f32)
  return __builtin_amdgcn_wmma_f32_16x16x32_f16(
      /*neg_a=*/false, a, /*neg_b=*/false, b,
      /*c_mod=*/(short)0, c, /*reuse_a=*/false, /*reuse_b=*/false);
}

// A-matrix 16x32 f16 fragment from row-major LDS tile (stride in halfs).
// Lane l: M = l&15, h = l>>4; k(e) = e + 16h
__device__ __forceinline__ v16h load_a_frag(const _Float16* base, int stride, int lane) {
  const int m = lane & 15;
  const int h = lane >> 4;
  const _Float16* row = base + (size_t)m * stride + 16 * h;
  v16h a;
#pragma unroll
  for (int e = 0; e < 16; ++e) {
    const int k = e;
    a[e] = row[k];
  }
  return a;
}

// B-matrix 32x16 fragment where the source is stored [N][K] row-major in LDS:
// B[k][n] = Mnk[(n0+n)*ldk + kbase + k].  Per lane this is 16 CONTIGUOUS
// halfs (32 B) -> lowers to 2x ds_load_b128.
__device__ __forceinline__ v16h load_b_frag_nk(const _Float16* Mnk, int ldk,
                                               int n0, int kbase, int lane) {
  const int n = lane & 15;
  const int h = lane >> 4;
  const _Float16* p = Mnk + (size_t)(n0 + n) * ldk + kbase + 16 * h;
  v16h b;
#pragma unroll
  for (int e = 0; e < 16; ++e) b[e] = p[e];
  return b;
}

// ---------------------------------------------------------------------------
// Kernel 1/2: fused LayerNorm + X @ W^T  (f32 in, f16 out)
// Block: 256 threads (8 waves), tile = 128 rows x NOUT cols, K = 128.
// ---------------------------------------------------------------------------
template <int NOUT>
__global__ __launch_bounds__(256) void ln_gemm_kernel(
    const float* __restrict__ X, const float* __restrict__ lnw,
    const float* __restrict__ lnb, const float* __restrict__ W,
    _Float16* __restrict__ out) {
  __shared__ _Float16 Alds[128 * 128];  // 32 KB, A[m][k]
  __shared__ _Float16 Wt[128 * 128];    // 32 KB, W rows as-is: [n][k]

  const int tid  = threadIdx.x;
  const int lane = tid & 31;
  const int wave = tid >> 5;
  const long m0  = (long)blockIdx.x * 128;

  // ---- LayerNorm: each wave handles 16 rows; 32 lanes x float4 = 128 elems
  const float4 w4 = ((const float4*)lnw)[lane];
  const float4 b4 = ((const float4*)lnb)[lane];
#pragma unroll 1
  for (int i = 0; i < 16; ++i) {
    const int rloc = wave * 16 + i;
    const float4 x = ((const float4*)(X + (size_t)(m0 + rloc) * 128))[lane];
    float s = x.x + x.y + x.z + x.w;
#pragma unroll
    for (int o = 1; o < 32; o <<= 1) s += __shfl_xor(s, o, 32);
    const float mean = s * (1.0f / 128.0f);
    const float dx = x.x - mean, dy = x.y - mean, dz = x.z - mean, dw = x.w - mean;
    float v = dx * dx + dy * dy + dz * dz + dw * dw;
#pragma unroll
    for (int o = 1; o < 32; o <<= 1) v += __shfl_xor(v, o, 32);
    const float rstd = rsqrtf(v * (1.0f / 128.0f) + 1e-5f);
    _Float16* dst = Alds + rloc * 128 + lane * 4;
    dst[0] = (_Float16)(dx * rstd * w4.x + b4.x);
    dst[1] = (_Float16)(dy * rstd * w4.y + b4.y);
    dst[2] = (_Float16)(dz * rstd * w4.z + b4.z);
    dst[3] = (_Float16)(dw * rstd * w4.w + b4.w);
  }

  v16h a[4];
  bool a_loaded = false;
#pragma unroll 1
  for (int half = 0; half < NOUT / 128; ++half) {
    // straight f32->f16 copy of 128 weight rows (no transpose needed: [n][k])
    for (int idx = tid; idx < 128 * 128; idx += 256)
      Wt[idx] = (_Float16)W[(size_t)half * 16384 + idx];
    __syncthreads();
    if (!a_loaded) {
#pragma unroll
      for (int kc = 0; kc < 4; ++kc)
        a[kc] = load_a_frag(Alds + wave * 16 * 128 + kc * 32, 128, lane);
      a_loaded = true;
    }
    const int nl = lane & 15, h = lane >> 4;
#pragma unroll 1
    for (int t = 0; t < 8; ++t) {
      v8f acc = {};
#pragma unroll
      for (int kc = 0; kc < 4; ++kc) {
        v16h b = load_b_frag_nk(Wt, 128, t * 16, kc * 32, lane);
        acc = wmma_f32(a[kc], b, acc);
      }
      const long col = (long)half * 128 + t * 16 + nl;
#pragma unroll
      for (int r = 0; r < 8; ++r)
        out[(size_t)(m0 + wave * 16 + r + 8 * h) * NOUT + col] = (_Float16)acc[r];
    }
    __syncthreads();
  }
}

// ---------------------------------------------------------------------------
// Kernel 3: per-head attention, two-pass online softmax with score recompute.
// grid = (512 bh, 4 nh, 2 row-blocks); block = 256 threads (8 waves).
// Pass A: 16 QK^T WMMAs -> row max.  Pass B: recompute scores, write
// unnormalized exp() P-chunks into a wave-private 1 KB LDS slice (reuses this
// wave's own Q rows -> zero barriers), 16 PV WMMAs, row-sum l, O *= 1/l.
// ---------------------------------------------------------------------------
__global__ __launch_bounds__(256) void attn_kernel(
    const _Float16* __restrict__ qbuf,   // (131072, 128) f16
    const _Float16* __restrict__ kvbuf,  // (131072, 256) f16: [k | v]
    const float* __restrict__ rel_emb,   // (511, 4) f32
    _Float16* __restrict__ obuf) {       // (131072, 128) f16
  __shared__ _Float16 smem[4096 + 8192 + 8192];  // 42 KB total (+rel below)
  __shared__ float rel_lds[511];

  const int tid  = threadIdx.x;
  const int lane = tid & 31;
  const int wave = tid >> 5;
  const int bh = blockIdx.x, nh = blockIdx.y, zb = blockIdx.z;

  _Float16* Qs = smem;           // [128][32]  (later: per-wave P chunks)
  _Float16* Ks = smem + 4096;    // [256][32]  K row-major (n, d)
  _Float16* Vt = smem + 12288;   // [32][256]  V transposed (d, kpos)

  const size_t row0 = (size_t)bh * 256;

  // ---- stage Q / K (dword copies) and V transposed (half scatter), + rel col
  {
    const unsigned int* qs = (const unsigned int*)qbuf;
    const unsigned int* ks = (const unsigned int*)kvbuf;
    unsigned int* qd = (unsigned int*)Qs;
    unsigned int* kd = (unsigned int*)Ks;
#pragma unroll 1
    for (int idx = tid; idx < 128 * 16; idx += 256) {  // Q: 128 rows x 16 dwords
      const int row = idx >> 4, p = idx & 15;
      qd[idx] = qs[(row0 + zb * 128 + row) * 64 + nh * 16 + p];
    }
#pragma unroll 1
    for (int idx = tid; idx < 256 * 16; idx += 256) {  // K: 256 rows x 16 dwords
      const int row = idx >> 4, p = idx & 15;
      kd[idx] = ks[(row0 + row) * 128 + nh * 16 + p];
    }
#pragma unroll 1
    for (int i = tid; i < 256 * 32; i += 256) {        // V -> Vt[d][kpos]
      const int kpos = i & 255, d = i >> 8;
      Vt[d * 256 + kpos] = kvbuf[(row0 + kpos) * 256 + 128 + nh * 32 + d];
    }
    for (int i = tid; i < 511; i += 256) rel_lds[i] = rel_emb[i * 4 + nh];
  }
  __syncthreads();

  const int nl = lane & 15, h = lane >> 4;
  const float scale = 0.17677669529663687f;  // 1/sqrt(32)
  // rel index for (r, tile t): relbase + t*16 - r
  const int relbase = nl - (zb * 128 + wave * 16 + 8 * h) + 255;

  const v16h aQ = load_a_frag(Qs + wave * 512, 32, lane);

  // ---- pass A: row maxima (per-lane, then 4-step merge across the 16 lanes)
  float m_r[8];
#pragma unroll
  for (int r = 0; r < 8; ++r) m_r[r] = -1e30f;
#pragma unroll
  for (int t = 0; t < 16; ++t) {
    v8f z = {};
    v16h bK = load_b_frag_nk(Ks, 32, t * 16, 0, lane);
    v8f s8 = wmma_f32(aQ, bK, z);
#pragma unroll
    for (int r = 0; r < 8; ++r) {
      const float s = s8[r] * scale + rel_lds[relbase + t * 16 - r];
      m_r[r] = fmaxf(m_r[r], s);
    }
  }
#pragma unroll
  for (int r = 0; r < 8; ++r)
#pragma unroll
    for (int o = 1; o < 16; o <<= 1) m_r[r] = fmaxf(m_r[r], __shfl_xor(m_r[r], o, 32));

  // ---- pass B: recompute scores, P chunks (16x32) in wave-private LDS,
  //      accumulate O (unnormalized) and row sums l.
  _Float16* Pw = Qs + wave * 512;  // this wave's own (dead) Q rows, 1 KB
  float l_r[8];
#pragma unroll
  for (int r = 0; r < 8; ++r) l_r[r] = 0.0f;
  v8f oacc[2] = {{}, {}};

#pragma unroll
  for (int c = 0; c < 8; ++c) {    // 32-key chunks
#pragma unroll
    for (int u = 0; u < 2; ++u) {  // two 16-key tiles per chunk
      const int t = c * 2 + u;
      v8f z = {};
      v16h bK = load_b_frag_nk(Ks, 32, t * 16, 0, lane);
      v8f s8 = wmma_f32(aQ, bK, z);
#pragma unroll
      for (int r = 0; r < 8; ++r) {
        const float s = s8[r] * scale + rel_lds[relbase + t * 16 - r];
        const float p = __expf(s - m_r[r]);
        l_r[r] += p;
        Pw[(r + 8 * h) * 32 + u * 16 + nl] = (_Float16)p;
      }
    }
    // DS ops are in-order within a wave: the P writes above are seen by these
    // loads, and next chunk's writes can't pass these reads.
    v16h aP = load_a_frag(Pw, 32, lane);
#pragma unroll
    for (int t2 = 0; t2 < 2; ++t2) {
      v16h bV = load_b_frag_nk(Vt, 256, t2 * 16, c * 32, lane);
      oacc[t2] = wmma_f32(aP, bV, oacc[t2]);
    }
  }

  // ---- finalize: merge row sums across the 16-lane group, scale, store
#pragma unroll
  for (int r = 0; r < 8; ++r) {
#pragma unroll
    for (int o = 1; o < 16; o <<= 1) l_r[r] += __shfl_xor(l_r[r], o, 32);
    l_r[r] = 1.0f / l_r[r];
  }
#pragma unroll
  for (int t2 = 0; t2 < 2; ++t2)
#pragma unroll
    for (int r = 0; r < 8; ++r) {
      const size_t row = row0 + zb * 128 + wave * 16 + r + 8 * h;
      obuf[row * 128 + nh * 32 + t2 * 16 + nl] = (_Float16)(oacc[t2][r] * l_r[r]);
    }
}

// ---------------------------------------------------------------------------
// Kernel 4: out = attn_out @ Wo^T  (f16 in, f32 out)
// ---------------------------------------------------------------------------
__global__ __launch_bounds__(256) void gemm_out_kernel(
    const _Float16* __restrict__ Ain, const float* __restrict__ W,
    float* __restrict__ out) {
  __shared__ _Float16 Alds[128 * 128];
  __shared__ _Float16 Wt[128 * 128];  // Wo rows as-is: [n][k]
  const int tid = threadIdx.x, lane = tid & 31, wave = tid >> 5;
  const long m0 = (long)blockIdx.x * 128;

  const uint4* src = (const uint4*)(Ain + (size_t)m0 * 128);
  uint4* dstu = (uint4*)Alds;
#pragma unroll 1
  for (int idx = tid; idx < 128 * 128 / 8; idx += 256) dstu[idx] = src[idx];
  for (int idx = tid; idx < 128 * 128; idx += 256) Wt[idx] = (_Float16)W[idx];
  __syncthreads();

  v16h a[4];
#pragma unroll
  for (int kc = 0; kc < 4; ++kc)
    a[kc] = load_a_frag(Alds + wave * 16 * 128 + kc * 32, 128, lane);

  const int nl = lane & 15, h = lane >> 4;
#pragma unroll 1
  for (int t = 0; t < 8; ++t) {
    v8f acc = {};
#pragma unroll
    for (int kc = 0; kc < 4; ++kc) {
      v16h b = load_b_frag_nk(Wt, 128, t * 16, kc * 32, lane);
      acc = wmma_f32(a[kc], b, acc);
    }
#pragma unroll
    for (int r = 0; r < 8; ++r)
      out[(size_t)(m0 + wave * 16 + r + 8 * h) * 128 + t * 16 + nl] = acc[r];
  }
}

// ---------------------------------------------------------------------------
extern "C" void kernel_launch(void* const* d_in, const int* in_sizes, int n_in,
                              void* d_out, int out_size, void* d_ws, size_t ws_size,
                              hipStream_t stream) {
  (void)in_sizes; (void)n_in; (void)out_size; (void)ws_size;
  const float* pair = (const float*)d_in[0];
  const float* ctx  = (const float*)d_in[1];
  const float* qn_w = (const float*)d_in[2];
  const float* qn_b = (const float*)d_in[3];
  const float* kvn_w = (const float*)d_in[4];
  const float* kvn_b = (const float*)d_in[5];
  const float* Wq  = (const float*)d_in[6];
  const float* Wkv = (const float*)d_in[7];
  const float* Wo  = (const float*)d_in[8];
  const float* rel = (const float*)d_in[9];
  float* out = (float*)d_out;

  const size_t M = 2 * 256 * 256;                 // 131072 rows
  _Float16* qbuf    = (_Float16*)d_ws;            // M x 128 f16 (32 MB)
  _Float16* kvbuf   = qbuf + M * 128;             // M x 256 f16 (64 MB)
  _Float16* attnbuf = kvbuf + M * 256;            // M x 128 f16 (32 MB)

  ln_gemm_kernel<128><<<1024, 256, 0, stream>>>(pair, qn_w, qn_b, Wq, qbuf);
  ln_gemm_kernel<256><<<1024, 256, 0, stream>>>(ctx, kvn_w, kvn_b, Wkv, kvbuf);
  attn_kernel<<<dim3(512, 4, 2), 256, 0, stream>>>(qbuf, kvbuf, rel, attnbuf);
  gemm_out_kernel<<<1024, 256, 0, stream>>>(attnbuf, Wo, out);
}
